// Mask_Rcnn_2851858285268
// MI455X (gfx1250) — compile-verified, matrix operand order checked by reading, not run on hardware
//
#include <hip/hip_runtime.h>
#include <hip/hip_bf16.h>

typedef __attribute__((ext_vector_type(16))) __bf16 v16bf;
typedef __attribute__((ext_vector_type(2)))  __bf16 v2bf;
typedef __attribute__((ext_vector_type(8)))  float  v8f;

#define NUM_CLS 80
#define CCH 256
#define NROI 256
#define OUTP 7
#define KDIM (CCH * OUTP * OUTP)   // 12544
#define HID 1024
#define IMGSZ 1024.0f
#define MIN_SIZE 16.0f

// ---------------- fp32 -> packed bf16 (round to nearest even) ----------------
__device__ __forceinline__ unsigned int pk2bf(float a, float b) {
#if __has_builtin(__builtin_amdgcn_cvt_pk_bf16_f32)
  const v2bf v = __builtin_amdgcn_cvt_pk_bf16_f32(a, b);
  return __builtin_bit_cast(unsigned int, v);
#else
  const v2bf v = {(__bf16)a, (__bf16)b};   // fptrunc RNE; backend picks cvt op
  return __builtin_bit_cast(unsigned int, v);
#endif
}

// ---------------- init: zero accumulators, seed roi/valid ----------------
__global__ void init_kernel(const float* __restrict__ rois,
                            float* __restrict__ roi_cur,
                            float* __restrict__ valid,
                            float* __restrict__ acc /* pre_b+pre_s contiguous */) {
  int idx = blockIdx.x * blockDim.x + threadIdx.x;
  if (idx < 102400) acc[idx] = 0.0f;
  if (idx < NROI * 4) roi_cur[idx] = rois[idx];
  if (idx < NROI) valid[idx] = 1.0f;
}

// ---------------- ROI Align over FPN levels ----------------
__global__ __launch_bounds__(256) void pool_kernel(
    const float* __restrict__ P0, const float* __restrict__ P1,
    const float* __restrict__ P2, const float* __restrict__ P3,
    const float* __restrict__ roi_cur, float* __restrict__ pooled) {
  const int r = blockIdx.x;
  const int c = threadIdx.x;           // one channel per thread

  const float ry1 = roi_cur[r * 4 + 0];
  const float rx1 = roi_cur[r * 4 + 1];
  const float ry2 = roi_cur[r * 4 + 2];
  const float rx2 = roi_cur[r * 4 + 3];

  const float area = (ry2 - ry1 + 1.0f) * (rx2 - rx1 + 1.0f);
  float lf = floorf(4.0f + log2f(sqrtf(area) / 224.0f));
  lf = fminf(fmaxf(lf, 2.0f), 5.0f);
  const int lvl = (int)lf - 2;

  const float* feat;
  int H, W;
  float scale;
  if (lvl == 0)      { feat = P0; H = 256; W = 256; scale = 0.25f;    }
  else if (lvl == 1) { feat = P1; H = 128; W = 128; scale = 0.125f;   }
  else if (lvl == 2) { feat = P2; H = 64;  W = 64;  scale = 0.0625f;  }
  else               { feat = P3; H = 32;  W = 32;  scale = 0.03125f; }

  const float y1 = ry1 * scale;
  const float x1 = rx1 * scale;
  const float rh = fmaxf(ry2 * scale - y1, 1.0f);
  const float rw = fmaxf(rx2 * scale - x1, 1.0f);
  const float bh = rh / (float)OUTP;
  const float bw = rw / (float)OUTP;

  const float* base = feat + (size_t)c * H * W;
  const float Hf = (float)H, Wf = (float)W;

  for (int py = 0; py < OUTP; ++py) {
    for (int px = 0; px < OUTP; ++px) {
      float acc = 0.0f;
      #pragma unroll
      for (int dy = 0; dy < 2; ++dy) {
        #pragma unroll
        for (int dx = 0; dx < 2; ++dx) {
          const float gy = (float)py + ((float)dy + 0.5f) * 0.5f;
          const float gx = (float)px + ((float)dx + 0.5f) * 0.5f;
          const float yy = y1 + gy * bh;
          const float xx = x1 + gx * bw;
          if (yy > -1.0f && yy < Hf && xx > -1.0f && xx < Wf) {
            const float yc = fminf(fmaxf(yy, 0.0f), Hf - 1.0f);
            const float xc = fminf(fmaxf(xx, 0.0f), Wf - 1.0f);
            const float y0f = floorf(yc);
            const float x0f = floorf(xc);
            const float ly = yc - y0f, lx = xc - x0f;
            const int y0i = (int)y0f, x0i = (int)x0f;
            const int y1i = min(y0i + 1, H - 1);
            const int x1i = min(x0i + 1, W - 1);
            const float f00 = base[y0i * W + x0i];
            const float f01 = base[y0i * W + x1i];
            const float f10 = base[y1i * W + x0i];
            const float f11 = base[y1i * W + x1i];
            acc += f00 * (1.0f - ly) * (1.0f - lx) + f01 * (1.0f - ly) * lx +
                   f10 * ly * (1.0f - lx) + f11 * ly * lx;
          }
        }
      }
      pooled[(size_t)r * KDIM + c * (OUTP * OUTP) + py * OUTP + px] = acc * 0.25f;
    }
  }
}

// ---------------- WMMA bf16 split-K GEMM: Part[z] = A@B over K-chunk z --------
// A: [M,K] fp32 row-major (M mult of 128, K mult of 32, Kc mult of 32)
// B: [K,N] fp32 row-major. Block tile 128x64, 8 waves -> 32x32 each.
// Software-pipelined + double-buffered LDS: one barrier per K-step.
#define BM 128
#define BN 64
#define BK 32

__global__ __launch_bounds__(256) void fc_gemm(
    const float* __restrict__ A, const float* __restrict__ B,
    float* __restrict__ Part, int M, int K, int N, int Kc) {
  // fragment-order LDS: [buf][frag][lane][16 halves] -> contiguous 32B per lane
  __shared__ __attribute__((aligned(32))) unsigned short As[2][BM / 16][32][16];
  __shared__ __attribute__((aligned(32))) unsigned short Bs[2][BN / 16][32][16];

  const int tid  = threadIdx.x;
  const int lane = tid & 31;
  const int wave = tid >> 5;
  const int wm = wave >> 1;      // 0..3  (M direction)
  const int wn = wave & 1;       // 0..1  (N direction)
  const int m0 = blockIdx.y * BM;
  const int n0 = blockIdx.x * BN;
  const int kbeg = blockIdx.z * Kc;
  const int kend = min(K, kbeg + Kc);

  v8f acc[2][2] = {};

  float4 ra[4];                // A: 4 groups of 4-consecutive-K
  float2 rbv[4];               // B: 2 adjacent columns x 4 consecutive K rows

  // ---- loop-invariant tile coordinates ----
  const int a_row[4] = {(tid + 0) >> 3, (tid + 256) >> 3,
                        (tid + 512) >> 3, (tid + 768) >> 3};
  const int a_kb  = (tid & 7) * 4;
  const int a_ln  = ((a_kb >> 3) & 1) * 16;                 // lane-half from k[3]
  const int a_jb  = (a_kb & 7) + ((a_kb >= 16) ? 8 : 0);    // half-slot base
  const int b_cc2 = (tid & 31) * 2;                         // column pair
  const int b_kr  = (tid >> 5) * 4;                         // K-row group
  const int b_ln  = (b_cc2 & 15) + (((b_kr >> 3) & 1) * 16);
  const int b_jb  = (b_kr & 7) + ((b_kr >= 16) ? 8 : 0);
  const int b_fr  = b_cc2 >> 4;

  auto stage_load = [&](int k0) {
    #pragma unroll
    for (int it = 0; it < 4; ++it) {
      ra[it] = *(const float4*)(A + (size_t)(m0 + a_row[it]) * K + k0 + a_kb);
    }
    const int cc = n0 + b_cc2;
    const float* bp = B + (size_t)(k0 + b_kr) * N + cc;
    if (((N & 1) == 0) && (cc + 1 < N)) {
      #pragma unroll
      for (int i = 0; i < 4; ++i) rbv[i] = *(const float2*)(bp + (size_t)i * N);
    } else {
      #pragma unroll
      for (int i = 0; i < 4; ++i) {
        rbv[i].x = (cc < N)     ? bp[(size_t)i * N]     : 0.0f;
        rbv[i].y = (cc + 1 < N) ? bp[(size_t)i * N + 1] : 0.0f;
      }
    }
  };

  auto stage_store = [&](int buf) {
    #pragma unroll
    for (int it = 0; it < 4; ++it) {
      const int row = a_row[it];
      const uint2 pk = {pk2bf(ra[it].x, ra[it].y), pk2bf(ra[it].z, ra[it].w)};
      *(uint2*)&As[buf][row >> 4][(row & 15) + a_ln][a_jb] = pk;
    }
    const uint2 p0 = {pk2bf(rbv[0].x, rbv[1].x), pk2bf(rbv[2].x, rbv[3].x)};
    const uint2 p1 = {pk2bf(rbv[0].y, rbv[1].y), pk2bf(rbv[2].y, rbv[3].y)};
    *(uint2*)&Bs[buf][b_fr][b_ln + 0][b_jb] = p0;
    *(uint2*)&Bs[buf][b_fr][b_ln + 1][b_jb] = p1;
  };

  // prologue: tile 0 -> regs -> LDS buf 0
  stage_load(kbeg);
  stage_store(0);
  __syncthreads();

  int cur = 0;
  for (int k0 = kbeg; k0 < kend; k0 += BK) {
    const bool more = (k0 + BK) < kend;
    if (more) stage_load(k0 + BK);       // next tile's global loads in flight

    #pragma unroll
    for (int i = 0; i < 2; ++i) {
      #pragma unroll
      for (int j = 0; j < 2; ++j) {
        const v16bf a = *(const v16bf*)&As[cur][2 * wm + i][lane][0];
        const v16bf b = *(const v16bf*)&Bs[cur][2 * wn + j][lane][0];
        acc[i][j] = __builtin_amdgcn_wmma_f32_16x16x32_bf16(
            false, a, false, b, (short)0, acc[i][j], false, false);
      }
    }

    if (more) stage_store(cur ^ 1);      // write other buffer while this one is read
    __syncthreads();                     // single barrier per K-step
    cur ^= 1;
  }

  // store raw partial sums (bias/relu applied in deterministic reduce)
  float* out = Part + (size_t)blockIdx.z * M * N;
  const int rbase = (lane >= 16) ? 8 : 0;
  const int cl = lane & 15;
  #pragma unroll
  for (int i = 0; i < 2; ++i) {
    #pragma unroll
    for (int j = 0; j < 2; ++j) {
      const int col = n0 + wn * 32 + j * 16 + cl;
      if (col < N) {
        #pragma unroll
        for (int p = 0; p < 8; ++p) {
          const int row = m0 + wm * 32 + i * 16 + rbase + p;
          out[(size_t)row * N + col] = acc[i][j][p];
        }
      }
    }
  }
}

// ---------------- fixed-order split-K reduction + bias + relu ----------------
__global__ void reduce_kernel(const float* __restrict__ Part,
                              const float* __restrict__ bias,
                              float* __restrict__ Cout,
                              int M, int N, int S, int relu) {
  const int idx = blockIdx.x * blockDim.x + threadIdx.x;
  if (idx >= M * N) return;
  float sum = 0.0f;
  for (int s = 0; s < S; ++s) sum += Part[(size_t)s * M * N + idx];
  sum += bias[idx % N];
  if (relu) sum = fmaxf(sum, 0.0f);
  Cout[idx] = sum;
}

// ---------------- per-ROI softmax + bbox decode + cascade state update -------
__global__ __launch_bounds__(256) void post_kernel(
    const float* __restrict__ logits,  // [256,81]
    const float* __restrict__ loc,     // [256,324]
    float* __restrict__ roi_cur, float* __restrict__ valid,
    float* __restrict__ pre_b, float* __restrict__ pre_s, int stage) {
  const int r = threadIdx.x;
  if (r >= NROI) return;

  const float stds_all[3][4] = {
      {0.1f, 0.1f, 0.2f, 0.2f},
      {0.05f, 0.05f, 0.1f, 0.1f},
      {1.0f / 30.0f, 1.0f / 30.0f, 1.0f / 15.0f, 1.0f / 15.0f}};
  const float s0 = stds_all[stage][0], s1 = stds_all[stage][1];
  const float s2 = stds_all[stage][2], s3 = stds_all[stage][3];

  const float* lg = logits + (size_t)r * (NUM_CLS + 1);
  const float* lc = loc + (size_t)r * (NUM_CLS + 1) * 4;

  float mx = lg[0];
  for (int k = 1; k <= NUM_CLS; ++k) mx = fmaxf(mx, lg[k]);
  float sum = 0.0f;
  for (int k = 0; k <= NUM_CLS; ++k) sum += expf(lg[k] - mx);
  const float inv = 1.0f / sum;

  const float y1 = roi_cur[r * 4 + 0], x1 = roi_cur[r * 4 + 1];
  const float y2 = roi_cur[r * 4 + 2], x2 = roi_cur[r * 4 + 3];
  const float ch = y2 - y1, cw = x2 - x1;
  const float cy = y1 + ch * 0.5f, cx = x1 + cw * 0.5f;

  float bestsc = -1.0f;
  float bb0 = 0, bb1 = 0, bb2 = 0, bb3 = 0;

  for (int k = 0; k < NUM_CLS; ++k) {
    const float* l4 = lc + (k + 1) * 4;
    const float dy = l4[0] * s0, dx = l4[1] * s1;
    const float dh = l4[2] * s2, dw = l4[3] * s3;
    const float py = dy * ch + cy, px = dx * cw + cx;
    const float hh = expf(dh) * ch, ww = expf(dw) * cw;
    const float b0 = py - hh * 0.5f, b1 = px - ww * 0.5f;
    const float b2 = py + hh * 0.5f, b3 = px + ww * 0.5f;
    pre_b[(size_t)r * (NUM_CLS * 4) + k * 4 + 0] += b0;
    pre_b[(size_t)r * (NUM_CLS * 4) + k * 4 + 1] += b1;
    pre_b[(size_t)r * (NUM_CLS * 4) + k * 4 + 2] += b2;
    pre_b[(size_t)r * (NUM_CLS * 4) + k * 4 + 3] += b3;
    const float sc = expf(lg[k + 1] - mx) * inv;
    pre_s[(size_t)r * NUM_CLS + k] += sc;
    if (sc > bestsc) { bestsc = sc; bb0 = b0; bb1 = b1; bb2 = b2; bb3 = b3; }
  }

  const float n0 = fminf(fmaxf(bb0, 0.0f), IMGSZ);
  const float n1 = fminf(fmaxf(bb1, 0.0f), IMGSZ);
  const float n2 = fminf(fmaxf(bb2, 0.0f), IMGSZ);
  const float n3 = fminf(fmaxf(bb3, 0.0f), IMGSZ);
  roi_cur[r * 4 + 0] = n0; roi_cur[r * 4 + 1] = n1;
  roi_cur[r * 4 + 2] = n2; roi_cur[r * 4 + 3] = n3;
  if (stage < 2) {
    if ((n2 - n0) < MIN_SIZE || (n3 - n1) < MIN_SIZE) valid[r] = 0.0f;
  }
}

// ---------------- finalize: out = (pre_b/3*m, pre_s/3*m) ----------------
__global__ void finalize_kernel(const float* __restrict__ pre_b,
                                const float* __restrict__ pre_s,
                                const float* __restrict__ valid,
                                float* __restrict__ out) {
  const int idx = blockIdx.x * blockDim.x + threadIdx.x;
  const float third = 1.0f / 3.0f;
  if (idx < NROI * NUM_CLS * 4) {
    out[idx] = pre_b[idx] * third * valid[idx / (NUM_CLS * 4)];
  } else if (idx < NROI * NUM_CLS * 4 + NROI * NUM_CLS) {
    const int i2 = idx - NROI * NUM_CLS * 4;
    out[idx] = pre_s[i2] * third * valid[i2 / NUM_CLS];
  }
}

// helper: launch split-K GEMM + reduction
static void run_fc(const float* A, const float* B, const float* bias,
                   float* Cout, float* part, int M, int K, int N, int S,
                   int relu, hipStream_t stream) {
  const int Kc = K / S;                 // all uses divide exactly; Kc % 32 == 0
  dim3 grid((N + BN - 1) / BN, M / BM, S);
  fc_gemm<<<grid, 256, 0, stream>>>(A, B, part, M, K, N, Kc);
  const int total = M * N;
  reduce_kernel<<<(total + 255) / 256, 256, 0, stream>>>(part, bias, Cout,
                                                         M, N, S, relu);
}

extern "C" void kernel_launch(void* const* d_in, const int* in_sizes, int n_in,
                              void* d_out, int out_size, void* d_ws, size_t ws_size,
                              hipStream_t stream) {
  const float* P0   = (const float*)d_in[0];
  const float* P1   = (const float*)d_in[1];
  const float* P2   = (const float*)d_in[2];
  const float* P3   = (const float*)d_in[3];
  const float* rois = (const float*)d_in[4];

  float* ws = (float*)d_ws;
  float* pooled  = ws;                            // 256*12544 = 3,211,264
  float* h1      = pooled + (size_t)NROI * KDIM;  // 262,144
  float* h2      = h1 + (size_t)NROI * HID;       // 262,144
  float* logits  = h2 + (size_t)NROI * HID;       // 256*81
  float* locbuf  = logits + (size_t)NROI * (NUM_CLS + 1);        // 256*324
  float* roi_cur = locbuf + (size_t)NROI * (NUM_CLS + 1) * 4;    // 1024
  float* valid   = roi_cur + (size_t)NROI * 4;                   // 256
  float* pre_b   = valid + NROI;                                 // 81,920
  float* pre_s   = pre_b + (size_t)NROI * NUM_CLS * 4;           // 20,480
  float* part    = pre_s + (size_t)NROI * NUM_CLS;               // 8*256*1024

  init_kernel<<<(102400 + 255) / 256, 256, 0, stream>>>(rois, roi_cur, valid, pre_b);

  for (int s = 0; s < 3; ++s) {
    const float* W1 = (const float*)d_in[5 + s * 8 + 0];
    const float* b1 = (const float*)d_in[5 + s * 8 + 1];
    const float* W2 = (const float*)d_in[5 + s * 8 + 2];
    const float* b2 = (const float*)d_in[5 + s * 8 + 3];
    const float* Wc = (const float*)d_in[5 + s * 8 + 4];
    const float* bc = (const float*)d_in[5 + s * 8 + 5];
    const float* Wl = (const float*)d_in[5 + s * 8 + 6];
    const float* bl = (const float*)d_in[5 + s * 8 + 7];

    pool_kernel<<<NROI, 256, 0, stream>>>(P0, P1, P2, P3, roi_cur, pooled);

    // FC1: [256,12544]@[12544,1024], split-K 8 -> 256 blocks
    run_fc(pooled, W1, b1, h1, part, NROI, KDIM, HID, 8, 1, stream);
    // FC2: [256,1024]@[1024,1024], split-K 4 -> 128 blocks
    run_fc(h1, W2, b2, h2, part, NROI, HID, HID, 4, 1, stream);
    // cls head: N=81, split-K 8 -> 32 blocks
    run_fc(h2, Wc, bc, logits, part, NROI, HID, NUM_CLS + 1, 8, 0, stream);
    // loc head: N=324, split-K 8 -> 96 blocks
    run_fc(h2, Wl, bl, locbuf, part, NROI, HID, (NUM_CLS + 1) * 4, 8, 0, stream);

    post_kernel<<<1, 256, 0, stream>>>(logits, locbuf, roi_cur, valid, pre_b, pre_s, s);
  }

  finalize_kernel<<<(102400 + 255) / 256, 256, 0, stream>>>(pre_b, pre_s, valid, (float*)d_out);
}
